// dot_attention_1735166788681
// MI455X (gfx1250) — matrix operand order repaired and checked by reading, not compiled, not run on hardware
//
#include <hip/hip_runtime.h>
#include <hip/hip_bf16.h>

typedef __attribute__((ext_vector_type(16))) __bf16 v16bf;
typedef __attribute__((ext_vector_type(8)))  float  v8f;

#define BATCH   16
#define SEQ     2048
#define DIM     128
#define ROWS    16               // query rows per workgroup
#define NTHR    256              // 8 wave32
#define KCHUNK  128              // K/V rows staged per pipeline stage
#define NCHUNKS (SEQ / KCHUNK)   // 16
#define KSTRIDE 132              // f32 dwords per staged row (pad -> bank spread)
#define QPAD    (DIM + 8)        // bf16 elems per Q row
#define SPAD    (SEQ + 4)        // f32 elems per score row (8208B: 16B aligned)

__device__ __forceinline__ void wait_async_zero() {
#if __has_builtin(__builtin_amdgcn_s_wait_asynccnt)
    __builtin_amdgcn_s_wait_asynccnt(0);
#else
    asm volatile("s_wait_asynccnt 0x0" ::: "memory");
#endif
}

__device__ __forceinline__ void async_copy_b128(unsigned lds_byte, const float* g) {
    unsigned long long ga = (unsigned long long)(size_t)g;
    asm volatile("global_load_async_to_lds_b128 %0, %1, off"
                 :: "v"(lds_byte), "v"(ga) : "memory");
}

// Async-stage 128 rows x 128 f32 (row stride DIM in global) into LDS buf
// (row stride KSTRIDE dwords). 4096 x b128 segments, 16 per thread, coalesced.
__device__ __forceinline__ void stage_chunk_async(const float* gsrc, const float* buf, int t) {
    unsigned base = (unsigned)(size_t)buf;   // low 32 bits of flat LDS addr = LDS offset
    #pragma unroll
    for (int i = 0; i < (KCHUNK * DIM / 4) / NTHR; ++i) {   // 16 issues/thread
        int idx = i * NTHR + t;
        int r   = idx >> 5;        // row 0..127
        int seg = idx & 31;        // 16B segment within row
        async_copy_b128(base + (unsigned)(r * KSTRIDE * 4 + seg * 16),
                        gsrc + (size_t)r * DIM + seg * 4);
    }
}

__global__ __launch_bounds__(NTHR)
void fused_attn_bf16_wmma(const float* __restrict__ q,
                          const float* __restrict__ k,
                          const float* __restrict__ v,
                          float* __restrict__ ctx,
                          float* __restrict__ attn)
{
    __shared__ __bf16 qs[ROWS * QPAD];               // ~4.25 KB
    __shared__ float  kbuf[2 * KCHUNK * KSTRIDE];    // ~132 KB double buffer (K, then V)
    __shared__ float  sc[ROWS * SPAD];               // ~128.25 KB score strip
    __shared__ float  invrow[ROWS];                  // per-row 1/sum

    const int blk  = blockIdx.x;
    const int b    = blk / (SEQ / ROWS);
    const int r0   = (blk % (SEQ / ROWS)) * ROWS;
    const int t    = threadIdx.x;
    const int lane = t & 31;
    const int w    = t >> 5;        // wave id 0..7
    const int half = lane >> 4;     // 0 or 1
    const int l16  = lane & 15;

    const float* qb = q + ((size_t)b * SEQ + r0) * DIM;
    const float* kb = k + (size_t)b * SEQ * DIM;
    const float* vb = v + (size_t)b * SEQ * DIM;

    float* buf0 = kbuf;
    float* buf1 = kbuf + KCHUNK * KSTRIDE;

    // kick off async copy of K chunk 0 immediately
    stage_chunk_async(kb, buf0, t);

    // ---- stage Q strip (16 x 128) as bf16 while K chunk 0 is in flight ----
    for (int i = t; i < ROWS * DIM; i += NTHR) {
        int m = i / DIM, d = i % DIM;
        qs[m * QPAD + d] = (__bf16)qb[(size_t)m * DIM + d];
    }
    wait_async_zero();
    __syncthreads();

    // ---- scores: S = Q K^T, double-buffered async K staging ----
    for (int jc = 0; jc < NCHUNKS; ++jc) {
        int j0 = jc * KCHUNK;
        if (jc + 1 < NCHUNKS)       // prefetch next chunk into the other buffer
            stage_chunk_async(kb + (size_t)(j0 + KCHUNK) * DIM,
                              (jc & 1) ? buf0 : buf1, t);
        const float* kcur = (jc & 1) ? buf1 : buf0;

        v8f c = {};
        #pragma unroll
        for (int dd = 0; dd < DIM; dd += 32) {
            v16bf a, bm;
            #pragma unroll
            for (int i = 0; i < 16; ++i) {
                int ak = dd + half * 16 + i;
                a[i]  = qs[l16 * QPAD + ak];                                  // A row l16
                bm[i] = (__bf16)kcur[(w * 16 + l16) * KSTRIDE + dd + half * 16 + i];
            }
            c = __builtin_amdgcn_wmma_f32_16x16x32_bf16(
                    false, a, false, bm, (short)0, c, false, false);
        }
        #pragma unroll
        for (int vi = 0; vi < 8; ++vi) {
            int m = vi + 8 * half;
            sc[m * SPAD + j0 + w * 16 + l16] = c[vi];
        }
        wait_async_zero();      // next chunk landed in LDS
        __syncthreads();        // all waves done reading current chunk + sc tile visible
    }

    // kbuf is free now: prefetch V chunk 0 so it flies during softmax
    stage_chunk_async(vb, buf0, t);

    // ---- softmax (wave w -> rows 2w, 2w+1), float4 passes ----
    // sc row ends up holding UNNORMALIZED e = exp(s - max); 1/sum goes to invrow[].
    #pragma unroll
    for (int rr = 0; rr < 2; ++rr) {
        int r = w * 2 + rr;
        float4* srow = reinterpret_cast<float4*>(sc + r * SPAD);

        float mx = -__builtin_inff();
        for (int i = lane; i < SEQ / 4; i += 32) {
            float4 x = srow[i];
            mx = fmaxf(mx, fmaxf(fmaxf(x.x, x.y), fmaxf(x.z, x.w)));
        }
        #pragma unroll
        for (int off = 16; off > 0; off >>= 1)
            mx = fmaxf(mx, __shfl_xor(mx, off, 32));

        float sum = 0.f;
        for (int i = lane; i < SEQ / 4; i += 32) {
            float4 x = srow[i];
            x.x = __expf(x.x - mx); x.y = __expf(x.y - mx);
            x.z = __expf(x.z - mx); x.w = __expf(x.w - mx);
            sum += (x.x + x.y) + (x.z + x.w);
            srow[i] = x;                       // keep e in LDS for the PV matmul
        }
        #pragma unroll
        for (int off = 16; off > 0; off >>= 1)
            sum += __shfl_xor(sum, off, 32);
        float inv = 1.0f / sum;
        if (lane == 0) invrow[r] = inv;

        float4* arow = reinterpret_cast<float4*>(attn + ((size_t)b * SEQ + (r0 + r)) * SEQ);
        for (int i = lane; i < SEQ / 4; i += 32) {
            float4 x = srow[i];
            x.x *= inv; x.y *= inv; x.z *= inv; x.w *= inv;
            arow[i] = x;                       // wide coalesced attention stream-out
        }
    }
    wait_async_zero();
    __syncthreads();

    // ---- context: C(16x128) = E(16x2048) x V(2048x128), async-pipelined V ----
    v8f cacc = {};
    for (int vc = 0; vc < NCHUNKS; ++vc) {
        if (vc + 1 < NCHUNKS)
            stage_chunk_async(vb + (size_t)(vc + 1) * KCHUNK * DIM,
                              (vc & 1) ? buf0 : buf1, t);
        const float* vcur = (vc & 1) ? buf1 : buf0;

        #pragma unroll
        for (int sub = 0; sub < 4; ++sub) {          // 4 x K=32 per 128-row chunk
            v16bf a, bm;
            #pragma unroll
            for (int i = 0; i < 16; ++i) {
                int ak = vc * KCHUNK + sub * 32 +
                         half * 16 + i;
                a[i]  = (__bf16)sc[l16 * SPAD + ak];                       // E row l16
                bm[i] = (__bf16)vcur[(sub * 32 + half * 16 + i) * KSTRIDE  // V row = K idx
                                     + w * 16 + l16];                      // col per lane
            }
            cacc = __builtin_amdgcn_wmma_f32_16x16x32_bf16(
                       false, a, false, bm, (short)0, cacc, false, false);
        }
        wait_async_zero();
        __syncthreads();
    }

    // deferred softmax normalization folded into the epilogue
    #pragma unroll
    for (int vi = 0; vi < 8; ++vi) {
        int m = vi + 8 * half;
        ctx[((size_t)b * SEQ + (r0 + m)) * DIM + w * 16 + l16] = cacc[vi] * invrow[m];
    }
}

extern "C" void kernel_launch(void* const* d_in, const int* in_sizes, int n_in,
                              void* d_out, int out_size, void* d_ws, size_t ws_size,
                              hipStream_t stream) {
    (void)in_sizes; (void)n_in; (void)d_ws; (void)ws_size; (void)out_size;
    const float* q = (const float*)d_in[0];
    const float* k = (const float*)d_in[1];
    const float* v = (const float*)d_in[2];
    float* ctx  = (float*)d_out;                                   // [B,S,D] first
    float* attn = ctx + (size_t)BATCH * SEQ * DIM;                 // then [B,S,S]

    dim3 grid(BATCH * (SEQ / ROWS));   // 2048 workgroups
    dim3 block(NTHR);
    fused_attn_bf16_wmma<<<grid, block, 0, stream>>>(q, k, v, ctx, attn);
}
